// TemporalEncoderRNN_89129161326947
// MI455X (gfx1250) — matrix operand order repaired and checked by reading, not compiled, not run on hardware
//
#include <hip/hip_runtime.h>
#include <hip/hip_bf16.h>
#include <cstdint>
#include <cstddef>

// Problem constants (match reference)
#define BB   256
#define TT   512
#define DIN  512
#define HH   1024
#define H3   3072
#define BT   (BB * TT)          // 131072
#define LN_EPS 1e-5f

typedef __attribute__((ext_vector_type(16))) __bf16 v16bf;
typedef __attribute__((ext_vector_type(8)))  float  v8f;

// ---------------------------------------------------------------------------
// WMMA fragment loaders (bf16, 16x16x32 shapes, wave32)
// Layouts per CDNA5 ISA §7.12.2.
// ---------------------------------------------------------------------------

// A-matrix 16x32 (MxK), row-major source with leading dim lda (elements).
// lane in [0,15]:  row=lane,    V0..3 = K[0..7],  V4..7 = K[16..23]
// lane in [16,31]: row=lane-16, V0..3 = K[8..15], V4..7 = K[24..31]
__device__ __forceinline__ v16bf load_a_frag(const __hip_bfloat16* __restrict__ A,
                                             int lda, int m0, int k0) {
  const int lane = threadIdx.x & 31;
  const int row  = m0 + (lane & 15);
  const int kh   = (lane >> 4) * 8;
  const __hip_bfloat16* p = A + (size_t)row * lda + k0 + kh;
  union { int4 i4[2]; v16bf v; } u;
  u.i4[0] = *(const int4*)(p);        // 8 bf16 = 16B
  u.i4[1] = *(const int4*)(p + 16);   // next K-half
  return u.v;
}

// B-matrix 32x16 (KxN) where B[k][n] = W[n][k], W row-major, leading dim ldw.
// lane in [0,15]:  col=lane,    V0..3 = K[0..7],  V4..7 = K[16..23]
// lane in [16,31]: col=lane-16, V0..3 = K[8..15], V4..7 = K[24..31]
__device__ __forceinline__ v16bf load_b_frag(const __hip_bfloat16* __restrict__ W,
                                             int ldw, int n0, int k0) {
  const int lane = threadIdx.x & 31;
  const int ncol = n0 + (lane & 15);
  const int kh   = (lane >> 4) * 8;
  const __hip_bfloat16* p = W + (size_t)ncol * ldw + k0 + kh;
  union { int4 i4[2]; v16bf v; } u;
  u.i4[0] = *(const int4*)(p);
  u.i4[1] = *(const int4*)(p + 16);
  return u.v;
}

#define WMMA_BF16(a, b, c) \
  __builtin_amdgcn_wmma_f32_16x16x32_bf16(false, (a), false, (b), (short)0, (c), false, false)

__device__ __forceinline__ float sigmoidf_(float x) { return 1.0f / (1.0f + __expf(-x)); }

// ---------------------------------------------------------------------------
// Utility kernels
// ---------------------------------------------------------------------------
__global__ __launch_bounds__(256) void cvt_f32_bf16_kernel(const float* __restrict__ src,
                                                           __hip_bfloat16* __restrict__ dst,
                                                           int n) {
  int i = blockIdx.x * 256 + threadIdx.x;
  if (i < n) dst[i] = __float2bfloat16(src[i]);
}

__global__ __launch_bounds__(256) void zero_h_kernel(float* __restrict__ hf,
                                                     __hip_bfloat16* __restrict__ hb, int n) {
  int i = blockIdx.x * 256 + threadIdx.x;
  if (i < n) { hf[i] = 0.0f; hb[i] = __float2bfloat16(0.0f); }
}

// ---------------------------------------------------------------------------
// Stage 1: x = tanh(in@Win^T + b_in + cbi) * sigmoid(in@Wgate^T + b_gate + cbg)
// One wave per 16x64 output strip; 4+4 f32 accumulators, A-frag reused 8x.
// ---------------------------------------------------------------------------
__global__ __launch_bounds__(32) void glu_gemm_kernel(
    const __hip_bfloat16* __restrict__ Xin,    // BT x DIN (bf16)
    const __hip_bfloat16* __restrict__ Win,    // HH x DIN (bf16)
    const __hip_bfloat16* __restrict__ Wgt,    // HH x DIN (bf16)
    const float* __restrict__ b_in, const float* __restrict__ b_gate,
    const float* __restrict__ cbi, const float* __restrict__ cbg,
    __hip_bfloat16* __restrict__ Xout)         // BT x HH (bf16)
{
  const int m0   = blockIdx.x * 16;
  const int n0   = blockIdx.y * 64;
  const int bidx = m0 / TT;   // 16-row tiles never cross a batch boundary (512 % 16 == 0)

  v8f accx[4] = {};
  v8f accg[4] = {};

  for (int k0 = 0; k0 < DIN; k0 += 32) {
    v16bf a = load_a_frag(Xin, DIN, m0, k0);
#pragma unroll
    for (int j = 0; j < 4; ++j) {
      v16bf bi = load_b_frag(Win, DIN, n0 + 16 * j, k0);
      accx[j] = WMMA_BF16(a, bi, accx[j]);
      v16bf bg = load_b_frag(Wgt, DIN, n0 + 16 * j, k0);
      accg[j] = WMMA_BF16(a, bg, accg[j]);
    }
  }

  const int lane = threadIdx.x & 31;
  const int col  = lane & 15;
  const int mh   = (lane >> 4) * 8;
#pragma unroll
  for (int j = 0; j < 4; ++j) {
#pragma unroll
    for (int r = 0; r < 8; ++r) {
      int m = m0 + r + mh;
      int n = n0 + 16 * j + col;
      float vx = accx[j][r] + b_in[n]   + cbi[(size_t)bidx * HH + n];
      float vg = accg[j][r] + b_gate[n] + cbg[(size_t)bidx * HH + n];
      float xv = tanhf(vx) * sigmoidf_(vg);
      Xout[(size_t)m * HH + n] = __float2bfloat16(xv);
    }
  }
}

// ---------------------------------------------------------------------------
// Stage 2: gi = x @ W_ih^T + b_ih    (BT x 3072, stored bf16)
// ---------------------------------------------------------------------------
__global__ __launch_bounds__(32) void gi_gemm_kernel(
    const __hip_bfloat16* __restrict__ X,      // BT x HH (bf16)
    const __hip_bfloat16* __restrict__ Wih,    // H3 x HH (bf16)
    const float* __restrict__ b_ih,
    __hip_bfloat16* __restrict__ Gi)           // BT x H3 (bf16)
{
  const int m0 = blockIdx.x * 16;
  const int n0 = blockIdx.y * 64;

  v8f acc[4] = {};

  for (int k0 = 0; k0 < HH; k0 += 32) {
    v16bf a = load_a_frag(X, HH, m0, k0);
#pragma unroll
    for (int j = 0; j < 4; ++j) {
      v16bf b = load_b_frag(Wih, HH, n0 + 16 * j, k0);
      acc[j] = WMMA_BF16(a, b, acc[j]);
    }
  }

  const int lane = threadIdx.x & 31;
  const int col  = lane & 15;
  const int mh   = (lane >> 4) * 8;
#pragma unroll
  for (int j = 0; j < 4; ++j) {
#pragma unroll
    for (int r = 0; r < 8; ++r) {
      int m = m0 + r + mh;
      int n = n0 + 16 * j + col;
      Gi[(size_t)m * H3 + n] = __float2bfloat16(acc[j][r] + b_ih[n]);
    }
  }
}

// ---------------------------------------------------------------------------
// Stage 3: one GRU time step, fully fused.
// Wave owns a 16x16 h-tile; accumulates all three gate GEMMs (same A-frag of
// h, three B-frag streams from W_hh row-blocks), then gate math + mask.
// Double-buffered h (f32 master for blending, bf16 copy for next step's WMMA).
// ---------------------------------------------------------------------------
__global__ __launch_bounds__(32) void gru_step_kernel(
    const __hip_bfloat16* __restrict__ h_in_bf,  // BB x HH
    const float* __restrict__ h_in_f,            // BB x HH
    const __hip_bfloat16* __restrict__ Whh,      // H3 x HH (bf16)
    const float* __restrict__ b_hh,              // H3
    const __hip_bfloat16* __restrict__ Gi,       // BT x H3 (bf16)
    const float* __restrict__ mask,              // BB x TT
    float* __restrict__ h_out_f,
    __hip_bfloat16* __restrict__ h_out_bf,
    float* __restrict__ Y,                       // BT x HH (f32, pre-LN)
    int t)
{
  const int m0 = blockIdx.x * 16;   // batch tile
  const int n0 = blockIdx.y * 16;   // hidden tile

  v8f ar = {}, az = {}, an = {};

  for (int k0 = 0; k0 < HH; k0 += 32) {
    v16bf a  = load_a_frag(h_in_bf, HH, m0, k0);
    v16bf br = load_b_frag(Whh, HH, n0,            k0);
    ar = WMMA_BF16(a, br, ar);
    v16bf bz = load_b_frag(Whh, HH, n0 + HH,       k0);
    az = WMMA_BF16(a, bz, az);
    v16bf bn = load_b_frag(Whh, HH, n0 + 2 * HH,   k0);
    an = WMMA_BF16(a, bn, an);
  }

  const int lane = threadIdx.x & 31;
  const int col  = lane & 15;
  const int mh   = (lane >> 4) * 8;
#pragma unroll
  for (int r = 0; r < 8; ++r) {
    int m = m0 + r + mh;            // batch index
    int n = n0 + col;               // hidden index
    size_t row = (size_t)m * TT + t;

    float gir = __bfloat162float(Gi[row * H3 + n]);
    float giz = __bfloat162float(Gi[row * H3 + n + HH]);
    float gin = __bfloat162float(Gi[row * H3 + n + 2 * HH]);

    float hr = ar[r] + b_hh[n];
    float hz = az[r] + b_hh[n + HH];
    float hn = an[r] + b_hh[n + 2 * HH];

    float rg = sigmoidf_(gir + hr);
    float zg = sigmoidf_(giz + hz);
    float ng = tanhf(gin + rg * hn);

    float hold = h_in_f[(size_t)m * HH + n];
    float hnew = (1.0f - zg) * ng + zg * hold;
    float hfin = (mask[row] > 0.5f) ? hnew : hold;

    h_out_f[(size_t)m * HH + n]  = hfin;
    h_out_bf[(size_t)m * HH + n] = __float2bfloat16(hfin);
    Y[row * HH + n] = hfin;
  }
}

// ---------------------------------------------------------------------------
// Stage 4: LayerNorm over H=1024, in place on d_out. One block (256 thr) / row.
// ---------------------------------------------------------------------------
__global__ __launch_bounds__(256) void layernorm_kernel(
    float* __restrict__ Y, const float* __restrict__ g, const float* __restrict__ b)
{
  __shared__ float s1[256];
  __shared__ float s2[256];
  const size_t row = blockIdx.x;
  float* p = Y + row * HH;
  const int tid = threadIdx.x;

  float4 v = ((const float4*)p)[tid];
  float s  = v.x + v.y + v.z + v.w;
  float ss = v.x * v.x + v.y * v.y + v.z * v.z + v.w * v.w;
  s1[tid] = s; s2[tid] = ss;
  __syncthreads();
  for (int off = 128; off > 0; off >>= 1) {
    if (tid < off) { s1[tid] += s1[tid + off]; s2[tid] += s2[tid + off]; }
    __syncthreads();
  }
  float mean = s1[0] * (1.0f / HH);
  float var  = s2[0] * (1.0f / HH) - mean * mean;
  float inv  = rsqrtf(var + LN_EPS);

  float4 gg = ((const float4*)g)[tid];
  float4 bb = ((const float4*)b)[tid];
  v.x = (v.x - mean) * inv * gg.x + bb.x;
  v.y = (v.y - mean) * inv * gg.y + bb.y;
  v.z = (v.z - mean) * inv * gg.z + bb.z;
  v.w = (v.w - mean) * inv * gg.w + bb.w;
  ((float4*)p)[tid] = v;
}

// ---------------------------------------------------------------------------
// Host launcher
// ---------------------------------------------------------------------------
extern "C" void kernel_launch(void* const* d_in, const int* in_sizes, int n_in,
                              void* d_out, int out_size, void* d_ws, size_t ws_size,
                              hipStream_t stream) {
  const float* inputs  = (const float*)d_in[0];   // B,T,DIN
  const float* mask    = (const float*)d_in[1];   // B,T
  const float* cbi     = (const float*)d_in[2];   // B,H
  const float* cbg     = (const float*)d_in[3];   // B,H
  const float* W_in    = (const float*)d_in[4];   // H,DIN
  const float* b_in    = (const float*)d_in[5];   // H
  const float* W_gate  = (const float*)d_in[6];   // H,DIN
  const float* b_gate  = (const float*)d_in[7];   // H
  const float* W_ih    = (const float*)d_in[8];   // 3H,H
  const float* b_ih    = (const float*)d_in[9];   // 3H
  const float* W_hh    = (const float*)d_in[10];  // 3H,H
  const float* b_hh    = (const float*)d_in[11];  // 3H
  const float* ln_g    = (const float*)d_in[12];  // H
  const float* ln_b    = (const float*)d_in[13];  // H
  float* Y = (float*)d_out;                       // B,T,H (f32)

  // Workspace layout (256B aligned regions)
  char* ws = (char*)d_ws;
  size_t off = 0;
  auto carve = [&](size_t bytes) -> char* {
    char* p = ws + off;
    off += (bytes + 255) & ~(size_t)255;
    return p;
  };
  __hip_bfloat16* in_bf   = (__hip_bfloat16*)carve((size_t)BT * DIN * 2); // 128 MB
  __hip_bfloat16* x_bf    = (__hip_bfloat16*)carve((size_t)BT * HH  * 2); // 256 MB
  __hip_bfloat16* gi_bf   = (__hip_bfloat16*)carve((size_t)BT * H3  * 2); // 768 MB
  __hip_bfloat16* win_bf  = (__hip_bfloat16*)carve((size_t)HH * DIN * 2);
  __hip_bfloat16* wgt_bf  = (__hip_bfloat16*)carve((size_t)HH * DIN * 2);
  __hip_bfloat16* wih_bf  = (__hip_bfloat16*)carve((size_t)H3 * HH  * 2);
  __hip_bfloat16* whh_bf  = (__hip_bfloat16*)carve((size_t)H3 * HH  * 2);
  float*          h_f[2];
  __hip_bfloat16* h_b[2];
  h_f[0] = (float*)carve((size_t)BB * HH * 4);
  h_f[1] = (float*)carve((size_t)BB * HH * 4);
  h_b[0] = (__hip_bfloat16*)carve((size_t)BB * HH * 2);
  h_b[1] = (__hip_bfloat16*)carve((size_t)BB * HH * 2);

  // --- bf16 conversions of inputs / weights ---
  auto cvt = [&](const float* s, __hip_bfloat16* d, size_t n) {
    cvt_f32_bf16_kernel<<<dim3((unsigned)((n + 255) / 256)), dim3(256), 0, stream>>>(s, d, (int)n);
  };
  cvt(inputs, in_bf,  (size_t)BT * DIN);
  cvt(W_in,   win_bf, (size_t)HH * DIN);
  cvt(W_gate, wgt_bf, (size_t)HH * DIN);
  cvt(W_ih,   wih_bf, (size_t)H3 * HH);
  cvt(W_hh,   whh_bf, (size_t)H3 * HH);

  // --- h0 = 0 ---
  {
    int n = BB * HH;
    zero_h_kernel<<<dim3((n + 255) / 256), dim3(256), 0, stream>>>(h_f[0], h_b[0], n);
  }

  // --- Stage 1: GLU front-end ---
  glu_gemm_kernel<<<dim3(BT / 16, HH / 64), dim3(32), 0, stream>>>(
      in_bf, win_bf, wgt_bf, b_in, b_gate, cbi, cbg, x_bf);

  // --- Stage 2: gi = x @ W_ih^T + b_ih ---
  gi_gemm_kernel<<<dim3(BT / 16, H3 / 64), dim3(32), 0, stream>>>(
      x_bf, wih_bf, b_ih, gi_bf);

  // --- Stage 3: sequential GRU scan, one fused kernel per step ---
  for (int t = 0; t < TT; ++t) {
    int cur = t & 1, nxt = cur ^ 1;
    gru_step_kernel<<<dim3(BB / 16, HH / 16), dim3(32), 0, stream>>>(
        h_b[cur], h_f[cur], whh_bf, b_hh, gi_bf, mask,
        h_f[nxt], h_b[nxt], Y, t);
  }

  // --- Stage 4: LayerNorm in place on d_out ---
  layernorm_kernel<<<dim3(BT), dim3(256), 0, stream>>>(Y, ln_g, ln_b);

  (void)in_sizes; (void)n_in; (void)out_size; (void)ws_size;
}
